// Attention_9491877724899
// MI455X (gfx1250) — matrix operand order repaired and checked by reading, not compiled, not run on hardware
//
#include <hip/hip_runtime.h>

// ---------------------------------------------------------------------------
// MHA block for MI455X (gfx1250, wave32, WMMA).
// bf16 operands / f32 accumulate via v_wmma_f32_16x16x32_bf16.
// Tiles staged LDS-side with GLOBAL_LOAD_ASYNC_TO_LDS_B128 (ASYNCcnt),
// double-buffered ping-pong pipeline.
//
// Shapes: B=2, N=2048, C=1024, H=16, d=64.  SCALE = 0.125.
// ---------------------------------------------------------------------------

typedef __bf16  v16bf __attribute__((ext_vector_type(16)));
typedef float   v8f   __attribute__((ext_vector_type(8)));
typedef int     vec4i __attribute__((vector_size(16)));   // matches builtin V4i

#if defined(__has_builtin)
#  if __has_builtin(__builtin_amdgcn_global_load_async_to_lds_b128) && \
      __has_builtin(__builtin_amdgcn_s_wait_asynccnt)
#    define HAVE_ASYNC 1
#  endif
#endif
#ifndef HAVE_ASYNC
#  define HAVE_ASYNC 0
#endif

#if HAVE_ASYNC
#  define ASYNC_WAIT(n) __builtin_amdgcn_s_wait_asynccnt(n)
#else
#  define ASYNC_WAIT(n)
#endif

__device__ __forceinline__ unsigned short f2bf(float f) {
  return __builtin_bit_cast(unsigned short, (__bf16)f);
}

// Copy 16 bytes global -> LDS.  Async (no VGPR round trip) when available.
__device__ __forceinline__ void async_copy16(unsigned short* lds_dst,
                                             const unsigned short* gsrc) {
#if HAVE_ASYNC
  __builtin_amdgcn_global_load_async_to_lds_b128(
      (__attribute__((address_space(1))) vec4i*)(const_cast<unsigned short*>(gsrc)),
      (__attribute__((address_space(3))) vec4i*)(lds_dst), 0, 0);
#else
  *(uint4*)lds_dst = *(const uint4*)gsrc;
#endif
}

// A-operand (16x32 bf16, MxK): lane = row M (lane%16); 16 contiguous K at
// (lane/16)*16.  p is pre-offset by (lane/16)*16.
__device__ __forceinline__ v16bf ld_a_split(const unsigned short* p) {
  union { v16bf v; uint4 q[2]; } u;
  u.q[0] = *(const uint4*)(p);
  u.q[1] = *(const uint4*)(p + 8);
  return u.v;
}

// B-operand (32x16 bf16, KxN): lane = col N (lane%16); 16 contiguous K at
// offset (lane/16)*16 of a [N][K] row.  p is pre-offset.
__device__ __forceinline__ v16bf ld_b_contig(const unsigned short* p) {
  union { v16bf v; uint4 q[2]; } u;
  u.q[0] = *(const uint4*)(p);
  u.q[1] = *(const uint4*)(p + 8);
  return u.v;
}

__device__ __forceinline__ v8f wmma_bf16(v16bf a, v16bf b, v8f c) {
  return __builtin_amdgcn_wmma_f32_16x16x32_bf16(false, a, false, b,
                                                 (short)0, c, false, false);
}

// ---------------------------------------------------------------------------
// Pre-pass: f32 -> bf16 conversions (x) and transposed bf16 weights [N][K],
// so every later tile is a contiguous bf16 b128 async copy.
// ---------------------------------------------------------------------------
__global__ __launch_bounds__(256) void cvt_bf16_kernel(
    const float* __restrict__ src, unsigned short* __restrict__ dst, int n) {
  const int i = blockIdx.x * 256 + threadIdx.x;
  if (i < n) dst[i] = f2bf(src[i]);
}

// src[rows][cols] (f32) -> dst[cols][rows] (bf16)
__global__ __launch_bounds__(256) void cvt_transpose_kernel(
    const float* __restrict__ src, unsigned short* __restrict__ dst,
    int rows, int cols) {
  const int i = blockIdx.x * 256 + threadIdx.x;
  if (i < rows * cols) {
    const int r = i / cols, c = i % cols;
    dst[(size_t)c * rows + r] = f2bf(src[i]);
  }
}

// ---------------------------------------------------------------------------
// Kernel 1: QKV GEMM.  xbf[4096,1024] @ wt[3072,1024]^T + bias.
// Block: 256 thr = 8 waves.  Tile M=128, N=64; each wave owns a 16x64 strip
// (4 accumulators, one A operand reused across 4 WMMAs per K-step).
// Epilogue scatters Q,K -> [B,H,N,64] bf16, V -> [B,H,64,N] bf16 (transposed).
// ---------------------------------------------------------------------------
__global__ __launch_bounds__(256) void qkv_gemm_kernel(
    const unsigned short* __restrict__ xbf, const unsigned short* __restrict__ wt,
    const float* __restrict__ bias,
    unsigned short* __restrict__ oq, unsigned short* __restrict__ ok,
    unsigned short* __restrict__ ovt)
{
  __shared__ alignas(16) unsigned short Abuf[2][128 * 40];  // 128 rows x 32 K
  __shared__ alignas(16) unsigned short Btbuf[2][64 * 32];  // 64 cols x 32 K

  const int tid  = threadIdx.x;
  const int lane = tid & 31, col = lane & 15, half = lane >> 4;
  const int wave = tid >> 5;                     // 0..7 -> M sub-row
  const int m0 = blockIdx.y * 128;
  const int n0 = blockIdx.x * 64;

  auto stage = [&](int buf, int k0) {
    #pragma unroll
    for (int t = 0; t < 2; ++t) {                // A: 128 rows x 4 chunks
      const int cid = tid + t * 256;
      const int row = cid >> 2, coff = (cid & 3) * 8;
      async_copy16(&Abuf[buf][row * 40 + coff],
                   xbf + (size_t)(m0 + row) * 1024 + k0 + coff);
    }
    {                                            // B: 64 rows x 4 chunks
      const int row = tid >> 2, coff = (tid & 3) * 8;
      async_copy16(&Btbuf[buf][row * 32 + coff],
                   wt + (size_t)(n0 + row) * 1024 + k0 + coff);
    }
  };

  v8f zero = {0.f, 0.f, 0.f, 0.f, 0.f, 0.f, 0.f, 0.f};
  v8f acc[4] = {zero, zero, zero, zero};

  stage(0, 0);                                   // 3 async instr/wave in flight
  for (int kt = 0; kt < 32; ++kt) {
    const int buf = kt & 1;
    if (kt + 1 < 32) { stage(buf ^ 1, (kt + 1) * 32); ASYNC_WAIT(3); }
    else             { ASYNC_WAIT(0); }
    __syncthreads();
    v16bf a = ld_a_split(&Abuf[buf][(wave * 16 + col) * 40 + half * 16]);
    #pragma unroll
    for (int nn = 0; nn < 4; ++nn) {
      v16bf b = ld_b_contig(&Btbuf[buf][(nn * 16 + col) * 32 + half * 16]);
      acc[nn] = wmma_bf16(a, b, acc[nn]);
    }
    __syncthreads();
  }

  #pragma unroll
  for (int nn = 0; nn < 4; ++nn) {
    const int cg = n0 + nn * 16 + col;
    const float bv = bias[cg];
    #pragma unroll
    for (int r = 0; r < 8; ++r) {
      const int m  = m0 + wave * 16 + r + half * 8;   // token 0..4095
      const int bi = m >> 11, ni = m & 2047;
      const unsigned short val = f2bf(acc[nn][r] + bv);
      if (cg < 1024) {
        const int h = cg >> 6, dc = cg & 63;
        oq[(((size_t)(bi * 16 + h) * 2048) + ni) * 64 + dc] = val;
      } else if (cg < 2048) {
        const int h = (cg - 1024) >> 6, dc = cg & 63;
        ok[(((size_t)(bi * 16 + h) * 2048) + ni) * 64 + dc] = val;
      } else {
        const int h = (cg - 2048) >> 6, dc = cg & 63;
        ovt[((size_t)(bi * 16 + h) * 64 + dc) * 2048 + ni] = val;
      }
    }
  }
}

// ---------------------------------------------------------------------------
// Kernel 2: flash attention.  One block per (b*h, 64-query tile); 4 waves x
// 16 queries.  K/V^T tiles (32 keys) double-buffered via async-to-LDS.
// S = Q K^T (4 WMMAs), online softmax (width-16 shfl reductions),
// O += P V (4 WMMAs).
// ---------------------------------------------------------------------------
__global__ __launch_bounds__(128) void attn_kernel(
    const unsigned short* __restrict__ q, const unsigned short* __restrict__ k,
    const unsigned short* __restrict__ vt, unsigned short* __restrict__ o)
{
  __shared__ alignas(16) unsigned short Ks[2][32 * 64];   // keys x dims
  __shared__ alignas(16) unsigned short Vts[2][64 * 32];  // dims x keys
  __shared__ alignas(16) unsigned short Ps[4][16 * 32];   // per-wave P

  const int tid  = threadIdx.x;
  const int lane = tid & 31, col = lane & 15, half = lane >> 4;
  const int w    = tid >> 5;
  const int bh   = blockIdx.x;                  // b*16 + h
  const int q0   = blockIdx.y * 64;
  const unsigned short* kptr  = k  + (size_t)bh * 2048 * 64;
  const unsigned short* vtptr = vt + (size_t)bh * 64 * 2048;

  // Q rows for this wave in registers: two A-operands (dims 0..31 / 32..63)
  const unsigned short* qrow =
      q + (size_t)bh * 2048 * 64 + (size_t)(q0 + w * 16 + col) * 64;
  const v16bf aQ0 = ld_a_split(qrow + half * 16);
  const v16bf aQ1 = ld_a_split(qrow + 32 + half * 16);

  auto stage = [&](int buf, int k0) {
    #pragma unroll
    for (int t = 0; t < 2; ++t) {               // K tile: 32 x 64
      const int cid = tid + t * 128;
      const int row = cid >> 3, coff = (cid & 7) * 8;
      async_copy16(&Ks[buf][row * 64 + coff],
                   kptr + (size_t)(k0 + row) * 64 + coff);
    }
    #pragma unroll
    for (int t = 0; t < 2; ++t) {               // V^T tile: 64 x 32
      const int cid = tid + t * 128;
      const int row = cid >> 2, coff = (cid & 3) * 8;
      async_copy16(&Vts[buf][row * 32 + coff],
                   vtptr + (size_t)row * 2048 + k0 + coff);
    }
  };

  float m_i[8], l_i[8];
  v8f zero = {0.f, 0.f, 0.f, 0.f, 0.f, 0.f, 0.f, 0.f};
  v8f co[4] = {zero, zero, zero, zero};
  #pragma unroll
  for (int r = 0; r < 8; ++r) { m_i[r] = -3.0e38f; l_i[r] = 0.f; }

  stage(0, 0);                                  // 4 async instr/wave in flight
  for (int it = 0; it < 64; ++it) {
    const int k0  = it * 32;
    const int buf = it & 1;
    if (it + 1 < 64) { stage(buf ^ 1, k0 + 32); ASYNC_WAIT(4); }
    else             { ASYNC_WAIT(0); }
    __builtin_prefetch(kptr + (size_t)((k0 + 64) & 2047) * 64 + lane * 64, 0, 1);
    __syncthreads();

    // S = Q K^T : two 16-key subtiles x (K=64 -> 2 WMMAs)
    v8f s0 = zero, s1 = zero;
    {
      v16bf b00 = ld_b_contig(&Ks[buf][col * 64 + half * 16]);
      v16bf b01 = ld_b_contig(&Ks[buf][col * 64 + 32 + half * 16]);
      s0 = wmma_bf16(aQ0, b00, s0);
      s0 = wmma_bf16(aQ1, b01, s0);
      v16bf b10 = ld_b_contig(&Ks[buf][(16 + col) * 64 + half * 16]);
      v16bf b11 = ld_b_contig(&Ks[buf][(16 + col) * 64 + 32 + half * 16]);
      s1 = wmma_bf16(aQ0, b10, s1);
      s1 = wmma_bf16(aQ1, b11, s1);
    }

    // online softmax: each S row lives across 16 lanes
    #pragma unroll
    for (int r = 0; r < 8; ++r) {
      const float v0 = s0[r] * 0.125f;
      const float v1 = s1[r] * 0.125f;
      float mx = fmaxf(v0, v1);
      #pragma unroll
      for (int off = 8; off > 0; off >>= 1) mx = fmaxf(mx, __shfl_xor(mx, off, 16));
      const float mn   = fmaxf(m_i[r], mx);
      const float corr = __expf(m_i[r] - mn);
      const float p0   = __expf(v0 - mn);
      const float p1   = __expf(v1 - mn);
      float rs = p0 + p1;
      #pragma unroll
      for (int off = 8; off > 0; off >>= 1) rs += __shfl_xor(rs, off, 16);
      l_i[r] = l_i[r] * corr + rs;
      m_i[r] = mn;
      #pragma unroll
      for (int nn = 0; nn < 4; ++nn) co[nn][r] *= corr;
      // C-layout -> row-major [16][32] in LDS for A-layout reload
      Ps[w][(r + half * 8) * 32 + col]      = f2bf(p0);
      Ps[w][(r + half * 8) * 32 + 16 + col] = f2bf(p1);
    }

    v16bf aP = ld_a_split(&Ps[w][col * 32 + half * 16]);
    #pragma unroll
    for (int nn = 0; nn < 4; ++nn) {
      v16bf bV = ld_b_contig(&Vts[buf][(nn * 16 + col) * 32 + half * 16]);
      co[nn] = wmma_bf16(aP, bV, co[nn]);
    }
    __syncthreads();
  }

  // epilogue: O / l -> bf16 [B, N, C]
  const int b = bh >> 4, h = bh & 15;
  #pragma unroll
  for (int r = 0; r < 8; ++r) {
    const int token = q0 + w * 16 + r + half * 8;
    const float inv = 1.0f / l_i[r];
    #pragma unroll
    for (int nn = 0; nn < 4; ++nn) {
      o[((size_t)(b * 2048 + token)) * 1024 + h * 64 + nn * 16 + col] =
          f2bf(co[nn][r] * inv);
    }
  }
}

// ---------------------------------------------------------------------------
// Kernel 3: output projection.  ows[4096,1024](bf16) @ wt[1024,1024]^T + bias.
// Same tiling as kernel 1; f32 output.
// ---------------------------------------------------------------------------
__global__ __launch_bounds__(256) void proj_gemm_kernel(
    const unsigned short* __restrict__ a_bf, const unsigned short* __restrict__ wt,
    const float* __restrict__ bias, float* __restrict__ out)
{
  __shared__ alignas(16) unsigned short Abuf[2][128 * 40];
  __shared__ alignas(16) unsigned short Btbuf[2][64 * 32];

  const int tid  = threadIdx.x;
  const int lane = tid & 31, col = lane & 15, half = lane >> 4;
  const int wave = tid >> 5;
  const int m0 = blockIdx.y * 128;
  const int n0 = blockIdx.x * 64;

  auto stage = [&](int buf, int k0) {
    #pragma unroll
    for (int t = 0; t < 2; ++t) {
      const int cid = tid + t * 256;
      const int row = cid >> 2, coff = (cid & 3) * 8;
      async_copy16(&Abuf[buf][row * 40 + coff],
                   a_bf + (size_t)(m0 + row) * 1024 + k0 + coff);
    }
    {
      const int row = tid >> 2, coff = (tid & 3) * 8;
      async_copy16(&Btbuf[buf][row * 32 + coff],
                   wt + (size_t)(n0 + row) * 1024 + k0 + coff);
    }
  };

  v8f zero = {0.f, 0.f, 0.f, 0.f, 0.f, 0.f, 0.f, 0.f};
  v8f acc[4] = {zero, zero, zero, zero};

  stage(0, 0);
  for (int kt = 0; kt < 32; ++kt) {
    const int buf = kt & 1;
    if (kt + 1 < 32) { stage(buf ^ 1, (kt + 1) * 32); ASYNC_WAIT(3); }
    else             { ASYNC_WAIT(0); }
    __syncthreads();
    v16bf a = ld_a_split(&Abuf[buf][(wave * 16 + col) * 40 + half * 16]);
    #pragma unroll
    for (int nn = 0; nn < 4; ++nn) {
      v16bf b = ld_b_contig(&Btbuf[buf][(nn * 16 + col) * 32 + half * 16]);
      acc[nn] = wmma_bf16(a, b, acc[nn]);
    }
    __syncthreads();
  }

  #pragma unroll
  for (int nn = 0; nn < 4; ++nn) {
    const int cg = n0 + nn * 16 + col;
    const float bv = bias[cg];
    #pragma unroll
    for (int r = 0; r < 8; ++r) {
      const int m = m0 + wave * 16 + r + half * 8;
      out[(size_t)m * 1024 + cg] = acc[nn][r] + bv;
    }
  }
}

// ---------------------------------------------------------------------------
extern "C" void kernel_launch(void* const* d_in, const int* in_sizes, int n_in,
                              void* d_out, int out_size, void* d_ws, size_t ws_size,
                              hipStream_t stream) {
  const float* x      = (const float*)d_in[0];
  const float* w_qkv  = (const float*)d_in[1];
  const float* b_qkv  = (const float*)d_in[2];
  const float* w_proj = (const float*)d_in[3];
  const float* b_proj = (const float*)d_in[4];
  float* out = (float*)d_out;

  char* ws = (char*)d_ws;
  unsigned short* qws    = (unsigned short*)(ws);                             // 8 MB
  unsigned short* kws    = (unsigned short*)(ws + (size_t)8  * 1024 * 1024);  // 8 MB
  unsigned short* vtws   = (unsigned short*)(ws + (size_t)16 * 1024 * 1024);  // 8 MB
  unsigned short* ows    = (unsigned short*)(ws + (size_t)24 * 1024 * 1024);  // 8 MB
  unsigned short* xbf    = (unsigned short*)(ws + (size_t)32 * 1024 * 1024);  // 8 MB
  unsigned short* wtqkv  = (unsigned short*)(ws + (size_t)40 * 1024 * 1024);  // 6 MB
  unsigned short* wtproj = (unsigned short*)(ws + (size_t)46 * 1024 * 1024);  // 2 MB

  const int NX = 4096 * 1024;
  cvt_bf16_kernel<<<(NX + 255) / 256, 256, 0, stream>>>(x, xbf, NX);
  cvt_transpose_kernel<<<(1024 * 3072 + 255) / 256, 256, 0, stream>>>(
      w_qkv, wtqkv, 1024, 3072);
  cvt_transpose_kernel<<<(1024 * 1024 + 255) / 256, 256, 0, stream>>>(
      w_proj, wtproj, 1024, 1024);

  qkv_gemm_kernel<<<dim3(48, 32), 256, 0, stream>>>(xbf, wtqkv, b_qkv,
                                                    qws, kws, vtws);
  attn_kernel<<<dim3(32, 32), 128, 0, stream>>>(qws, kws, vtws, ows);
  proj_gemm_kernel<<<dim3(16, 32), 256, 0, stream>>>(ows, wtproj, b_proj, out);
}
